// RingAttentionDeepSpeed_76312978915791
// MI455X (gfx1250) — matrix operand order repaired and checked by reading, not compiled, not run on hardware
//
#include <hip/hip_runtime.h>

// Problem constants (match reference)
#define BN   4096
#define HH   8
#define DD   64
#define HD   (HH * DD)      // 512 floats between consecutive n rows
#define NBLK (BN / 32)      // 128 key blocks of 32

// LDS layout (bf16 tiles, padded for conflict-free strided gathers,
// strides are multiples of 16B so ds_load_b128 stays aligned):
//  K  tile: 32 rows (key)  x 64 bf16, row stride 144 B  (36 dw: 36*i distinct mod 64)
//  V^T tile: 64 rows (d)   x 32 bf16, row stride  80 B  (20 dw: 20*i distinct mod 64)
#define KSTRIDE 144
#define VSTRIDE 80
#define KBUF (32 * KSTRIDE)     // 4608 B
#define VBUF (64 * VSTRIDE)     // 5120 B
#define SMEM_BYTES (2 * KBUF + 2 * VBUF)   // 19456 B (epilogue reuses 17408 B)

typedef __bf16 bf16_t;
typedef __attribute__((ext_vector_type(16))) __bf16 v16bf;
typedef __attribute__((ext_vector_type(8)))  __bf16 v8bf;
typedef __attribute__((ext_vector_type(2)))  __bf16 v2bf;
typedef __attribute__((ext_vector_type(8)))  float  v8f;
typedef __attribute__((ext_vector_type(8)))  unsigned int v8u;

#define WMMA_BF16(A, B, C) \
  __builtin_amdgcn_wmma_f32_16x16x32_bf16(false, (A), false, (B), (short)0, (C), false, false)

// CDNA5 async store: LDS -> global, 16B, GVS mode, tracked by ASYNCcnt.
__device__ __forceinline__ void async_l2g_b128(unsigned goff, unsigned lds_off,
                                               const float* saddr) {
  asm volatile("global_store_async_from_lds_b128 %0, %1, %2 offset:0"
               :: "v"(goff), "v"(lds_off), "s"(saddr)
               : "memory");
}
__device__ __forceinline__ void wait_async0() {
  asm volatile("s_wait_asynccnt 0" ::: "memory");
}

__device__ __forceinline__ v8bf cvt8(float4 a, float4 b) {
  v8bf h;
  h[0] = (bf16_t)a.x; h[1] = (bf16_t)a.y; h[2] = (bf16_t)a.z; h[3] = (bf16_t)a.w;
  h[4] = (bf16_t)b.x; h[5] = (bf16_t)b.y; h[6] = (bf16_t)b.z; h[7] = (bf16_t)b.w;
  return h;
}
__device__ __forceinline__ v16bf cat16(v8bf lo, v8bf hi) {
  return __builtin_shufflevector(lo, hi, 0,1,2,3,4,5,6,7,8,9,10,11,12,13,14,15);
}

__global__ __launch_bounds__(128)
void fa_fwd_gfx1250(const float* __restrict__ q, const float* __restrict__ k,
                    const float* __restrict__ v, float* __restrict__ out) {
  __shared__ __align__(16) unsigned char smem[SMEM_BYTES];

  const int tid  = threadIdx.x;
  const int lane = tid & 31;
  const int wv   = tid >> 5;        // wave id 0..3
  const int ln   = lane & 15;       // row/col index within lane group
  const int g    = lane >> 4;       // lane group (0: lanes 0-15, 1: 16-31)

  const int head = blockIdx.x >> 6;       // 8 heads
  const int qblk = blockIdx.x & 63;       // 64 query blocks of 64 rows
  const int qb   = qblk * 64 + wv * 16;   // this wave's 16-query tile base

  // ---- Q fragments: B operand of S^T = K * Q^T. Fold (1/sqrt(D))*log2(e). ----
  // B(32x16) layout: lane holds column N=ln, K(=d) = 16*g + halfIndex.
  const float qscale = 0.125f * 1.44269504088896340736f;
  v16bf bq0, bq1;   // d = 0..31 and d = 32..63
  {
    const float* qp = q + ((size_t)(qb + ln) * HH + head) * DD + 16 * g;
    float4 t0 = *(const float4*)(qp + 0);
    float4 t1 = *(const float4*)(qp + 4);
    float4 t2 = *(const float4*)(qp + 8);
    float4 t3 = *(const float4*)(qp + 12);
    float4 u0 = *(const float4*)(qp + 32);
    float4 u1 = *(const float4*)(qp + 36);
    float4 u2 = *(const float4*)(qp + 40);
    float4 u3 = *(const float4*)(qp + 44);
#pragma unroll
    for (int i = 0; i < 4; ++i) {
      ((float*)&t0)[i] *= qscale; ((float*)&t1)[i] *= qscale;
      ((float*)&t2)[i] *= qscale; ((float*)&t3)[i] *= qscale;
      ((float*)&u0)[i] *= qscale; ((float*)&u1)[i] *= qscale;
      ((float*)&u2)[i] *= qscale; ((float*)&u3)[i] *= qscale;
    }
    bq0 = cat16(cvt8(t0, t1), cvt8(t2, t3));
    bq1 = cat16(cvt8(u0, u1), cvt8(u2, u3));
  }

  // ---- staging: convert fp32 -> bf16 once per tile (shared by 4 waves) ------
  auto stage = [&](int j, int buf) {
    const float* ksrc = k + (size_t)j * 32 * HD + (size_t)head * DD;
    const float* vsrc = v + (size_t)j * 32 * HD + (size_t)head * DD;
    unsigned char* kbuf = smem + buf * KBUF;
    unsigned char* vbuf = smem + 2 * KBUF + buf * VBUF;
    // K: 32x64 floats -> bf16 row-major. 256 8-float chunks / 128 threads.
#pragma unroll
    for (int i = 0; i < 2; ++i) {
      const int idx = tid + i * 128;
      const int r   = idx >> 3;          // key row
      const int c8  = idx & 7;           // 8-float chunk (coalesced 128B/8thr)
      const float* s = ksrc + r * HD + c8 * 8;
      float4 a = *(const float4*)s;
      float4 b = *(const float4*)(s + 4);
      *(v8bf*)(kbuf + r * KSTRIDE + c8 * 16) = cvt8(a, b);
    }
    // V: transpose to vT[d][key], bf16, key pairs packed into dwords.
#pragma unroll
    for (int i = 0; i < 2; ++i) {
      const int idx = tid + i * 128;
      const int rp  = idx >> 4;          // key pair -> keys 2rp, 2rp+1
      const int d4  = idx & 15;          // d chunk of 4 (coalesced reads)
      const float* s0 = vsrc + (2 * rp) * HD + d4 * 4;
      float4 a = *(const float4*)s0;           // row 2rp
      float4 b = *(const float4*)(s0 + HD);    // row 2rp+1
      unsigned char* dst = vbuf + rp * 4;      // byte offset 2*key
      v2bf p;
      p[0] = (bf16_t)a.x; p[1] = (bf16_t)b.x; *(v2bf*)(dst + (d4*4+0)*VSTRIDE) = p;
      p[0] = (bf16_t)a.y; p[1] = (bf16_t)b.y; *(v2bf*)(dst + (d4*4+1)*VSTRIDE) = p;
      p[0] = (bf16_t)a.z; p[1] = (bf16_t)b.z; *(v2bf*)(dst + (d4*4+2)*VSTRIDE) = p;
      p[0] = (bf16_t)a.w; p[1] = (bf16_t)b.w; *(v2bf*)(dst + (d4*4+3)*VSTRIDE) = p;
    }
  };

  // ---- flash-attention state ------------------------------------------------
  float m = -3.0e38f, l = 0.0f;
  v8f o[4] = {};                 // O^T tiles: d in [16t,16t+16), q = ln

  stage(0, 0);
  __syncthreads();

  for (int j = 0; j < NBLK; ++j) {
    const int buf = j & 1;
    if (j + 1 < NBLK) stage(j + 1, buf ^ 1);

    const unsigned char* kbuf = smem + buf * KBUF;
    const unsigned char* vbuf = smem + 2 * KBUF + buf * VBUF;

    // ---- S^T = K * Q^T : 4 WMMAs. A(16x32): lane row M=ln,
    // halves 0-15 -> K=16g+0..15 (byte 32g).
    const unsigned char* kr0 = kbuf + ln * KSTRIDE;          // keys 0..15
    const unsigned char* kr1 = kbuf + (16 + ln) * KSTRIDE;   // keys 16..31
    v16bf a00 = cat16(*(const v8bf*)(kr0 +      32*g), *(const v8bf*)(kr0 + 16 + 32*g));
    v16bf a01 = cat16(*(const v8bf*)(kr0 + 64 + 32*g), *(const v8bf*)(kr0 + 80 + 32*g));
    v16bf a10 = cat16(*(const v8bf*)(kr1 +      32*g), *(const v8bf*)(kr1 + 16 + 32*g));
    v16bf a11 = cat16(*(const v8bf*)(kr1 + 64 + 32*g), *(const v8bf*)(kr1 + 80 + 32*g));

    const v8f zero = {};
    v8f st0 = WMMA_BF16(a00, bq0, zero);
    st0     = WMMA_BF16(a01, bq1, st0);
    v8f st1 = WMMA_BF16(a10, bq0, zero);
    st1     = WMMA_BF16(a11, bq1, st1);

    // ---- online softmax (exp2 domain; scores already carry log2e) ----------
    float bm = -3.0e38f;
#pragma unroll
    for (int i = 0; i < 8; ++i) { bm = fmaxf(bm, st0[i]); bm = fmaxf(bm, st1[i]); }
    bm = fmaxf(bm, __shfl_xor(bm, 16, 32));
    const float mn   = fmaxf(m, bm);
    const float corr = __builtin_amdgcn_exp2f(m - mn);
    // vector sub with scalar broadcast -> packed/dual-issued f32 subs
    const v8f d0 = st0 - mn;
    const v8f d1 = st1 - mn;
    float p0[8], p1[8];
    v8f ps0, ps1;
#pragma unroll
    for (int i = 0; i < 8; ++i) {
      p0[i] = __builtin_amdgcn_exp2f(d0[i]);
      p1[i] = __builtin_amdgcn_exp2f(d1[i]);
      ps0[i] = p0[i]; ps1[i] = p1[i];
    }
    const v8f psum = ps0 + ps1;            // packed adds
    float ls = (((psum[0] + psum[1]) + (psum[2] + psum[3])) +
                ((psum[4] + psum[5]) + (psum[6] + psum[7])));
    ls += __shfl_xor(ls, 16, 32);
    l = l * corr + ls;
    m = mn;
    // accumulator rescale as whole-vector ops -> v_pk_mul_f32
    o[0] = o[0] * corr; o[1] = o[1] * corr;
    o[2] = o[2] * corr; o[3] = o[3] * corr;

    // ---- build P^T B-fragment: pack to bf16 dwords FIRST, then exchange ----
    // group 0 needs keys 0..15 (own pk0 | partner pk0), group 1 keys 16..31.
    unsigned pk0[4], pk1[4], pk0x[4], pk1x[4];
#pragma unroll
    for (int i = 0; i < 4; ++i) {
      v2bf c0; c0[0] = (bf16_t)p0[2*i]; c0[1] = (bf16_t)p0[2*i+1];
      v2bf c1; c1[0] = (bf16_t)p1[2*i]; c1[1] = (bf16_t)p1[2*i+1];
      pk0[i] = __builtin_bit_cast(unsigned, c0);
      pk1[i] = __builtin_bit_cast(unsigned, c1);
    }
#pragma unroll
    for (int i = 0; i < 4; ++i) {
      pk0x[i] = __shfl_xor(pk0[i], 16, 32);
      pk1x[i] = __shfl_xor(pk1[i], 16, 32);
    }
    v8u w;
#pragma unroll
    for (int i = 0; i < 4; ++i) {
      w[i]     = (g == 0) ? pk0[i]  : pk1x[i];
      w[4 + i] = (g == 0) ? pk0x[i] : pk1[i];
    }
    const v16bf pb = __builtin_bit_cast(v16bf, w);

    // ---- O^T += V^T * P^T : 4 WMMAs ----------------------------------------
#pragma unroll
    for (int t = 0; t < 4; ++t) {
      const unsigned char* vr = vbuf + (16 * t + ln) * VSTRIDE;  // row d=16t+ln
      v16bf av = cat16(*(const v8bf*)(vr + 32*g), *(const v8bf*)(vr + 16 + 32*g));
      o[t] = WMMA_BF16(av, pb, o[t]);
    }

    __syncthreads();   // all waves done reading buf; safe to overwrite next iter
  }

  // ---- epilogue: normalize -> LDS transpose -> coalesced async b128 stores --
  const float invl = 1.0f / l;
  o[0] = o[0] * invl; o[1] = o[1] * invl;
  o[2] = o[2] * invl; o[3] = o[3] * invl;
  float* lout = (float*)smem;                 // 64 rows x 68 floats (17408 B)
#pragma unroll
  for (int t = 0; t < 4; ++t)
#pragma unroll
    for (int i = 0; i < 8; ++i)
      lout[(wv * 16 + ln) * 68 + 16 * t + 8 * g + i] = o[t][i];
  __syncthreads();

  // 64 rows x 2 half-rows of 32 floats = 128 units, one per thread.
  {
    const int row  = tid >> 1;                // query row within block tile
    const int half = tid & 1;                 // d 0..31 / 32..63
    const float* obase = out + (size_t)(qblk * 64) * HD + (size_t)head * DD;
    unsigned goff   = (unsigned)(row * HD * 4 + half * 128);
    unsigned ldsoff = (unsigned)((row * 68 + half * 32) * 4);
#pragma unroll
    for (int i = 0; i < 8; ++i)
      async_l2g_b128(goff + i * 16u, ldsoff + i * 16u, obase);
    wait_async0();
  }
}

extern "C" void kernel_launch(void* const* d_in, const int* in_sizes, int n_in,
                              void* d_out, int out_size, void* d_ws, size_t ws_size,
                              hipStream_t stream) {
  const float* q = (const float*)d_in[0];
  const float* k = (const float*)d_in[1];
  const float* v = (const float*)d_in[2];
  float* out = (float*)d_out;
  (void)in_sizes; (void)n_in; (void)out_size; (void)d_ws; (void)ws_size;

  dim3 grid(HH * (BN / 64));   // 512 workgroups
  dim3 block(128);             // 4 waves, each owns a 16-query tile
  fa_fwd_gfx1250<<<grid, block, 0, stream>>>(q, k, v, out);
}
